// GCNNet_20770461844115
// MI455X (gfx1250) — matrix operand order repaired and checked by reading, not compiled, not run on hardware
//
#include <hip/hip_runtime.h>
#include <hip/hip_bf16.h>
#include <math.h>

typedef __attribute__((ext_vector_type(16))) _Float16 v16h;
typedef __attribute__((ext_vector_type(8)))  _Float16 v8h;
typedef __attribute__((ext_vector_type(8)))  float    v8f;

#define D_FEAT 256

// ---------------- helpers ----------------
__device__ __forceinline__ float lrelu(float x) { return x > 0.f ? x : 0.2f * x; }

// monotonic float<->uint key for atomicMax on floats of any sign
__device__ __forceinline__ unsigned fkey(float x) {
    unsigned u = __float_as_uint(x);
    return (u & 0x80000000u) ? ~u : (u | 0x80000000u);
}
__device__ __forceinline__ float unfkey(unsigned k) {
    unsigned u = (k & 0x80000000u) ? (k & 0x7fffffffu) : ~k;
    return __uint_as_float(u);
}
__device__ __forceinline__ float wave_sum32(float v) {
    for (int off = 16; off > 0; off >>= 1) v += __shfl_xor(v, off, 32);
    return v;
}
__device__ __forceinline__ float wave_max32(float v) {
    for (int off = 16; off > 0; off >>= 1) v = fmaxf(v, __shfl_xor(v, off, 32));
    return v;
}

// ---------------- generic utility kernels ----------------
__global__ void zero_f32(float* __restrict__ p, long long n) {
    long long i = (long long)blockIdx.x * blockDim.x + threadIdx.x;
    long long stride = (long long)gridDim.x * blockDim.x;
    for (; i < n; i += stride) p[i] = 0.f;
}

// ---------------- standardization ----------------
__global__ void col_stats(const float* __restrict__ x, float* __restrict__ csum,
                          float* __restrict__ csq, int N) {
    const int c = threadIdx.x;  // 256 threads = one per column
    float s = 0.f, s2 = 0.f;
    for (int r = blockIdx.x; r < N; r += gridDim.x) {
        float v = x[(size_t)r * D_FEAT + c];
        s += v; s2 += v * v;
    }
    atomicAdd(csum + c, s);
    atomicAdd(csq + c, s2);
}

__global__ void finalize_stats(const float* __restrict__ csum, const float* __restrict__ csq,
                               float* __restrict__ mean, float* __restrict__ rinv, int N) {
    int c = threadIdx.x;
    float m = csum[c] / (float)N;
    float var = (csq[c] - (float)N * m * m) / (float)(N - 1);  // ddof=1
    mean[c] = m;
    rinv[c] = rsqrtf(var);
}

__global__ void standardize(const float* __restrict__ x, const float* __restrict__ mean,
                            const float* __restrict__ rinv, _Float16* __restrict__ act,
                            long long total) {
    long long i = (long long)blockIdx.x * blockDim.x + threadIdx.x;
    if (i >= total) return;
    int c = (int)(i & (D_FEAT - 1));
    act[i] = (_Float16)((x[i] - mean[c]) * rinv[c]);
}

__global__ void convert_w(const float* __restrict__ w, _Float16* __restrict__ wh, int n) {
    int i = blockIdx.x * blockDim.x + threadIdx.x;
    if (i < n) wh[i] = (_Float16)w[i];
}

__global__ void convert_w2_pad(const float* __restrict__ w, _Float16* __restrict__ wh,
                               int K, int L, int Lp) {
    int i = blockIdx.x * blockDim.x + threadIdx.x;
    if (i >= K * Lp) return;
    int k = i / Lp, c = i - k * Lp;
    wh[i] = (c < L) ? (_Float16)w[k * L + c] : (_Float16)0.f;
}

// ---------------- WMMA GEMM: C[M,Nout] = A[M,K](f16) * B[K,Nout](f16) ----------------
// One wave -> 64x16 output slab (4 accumulators share each B fragment);
// K stepped by 32 (v_wmma_f32_16x16x32_f16). Tail slab clamps (overlap-recompute)
// so the inner loop has NO divergence and EXEC stays all-1s (ISA requirement).
// Requires M >= 16*MBLK (true here: M = 50000).
#define MBLK 4
__global__ void gemm_f16_wmma(const _Float16* __restrict__ A, const _Float16* __restrict__ B,
                              float* __restrict__ C, int M, int K, int Nout) {
    const int wave = blockIdx.x * (blockDim.x >> 5) + (threadIdx.x >> 5);
    const int lane = threadIdx.x & 31;
    const int tiles_m = M >> 4;
    const int tiles_n = Nout >> 4;
    const int mblocks = (tiles_m + MBLK - 1) / MBLK;
    if (wave >= mblocks * tiles_n) return;  // wave-uniform
    int mt0 = (wave % mblocks) * MBLK;
    if (mt0 + MBLK > tiles_m) mt0 = tiles_m - MBLK;  // clamp: overlapping tail slab
    const int n0 = (wave / mblocks) << 4;

    v8f acc[MBLK];
#pragma unroll
    for (int t = 0; t < MBLK; ++t) acc[t] = (v8f){};

    const int kb   = (lane < 16) ? 0 : 8;             // ISA 16-bit A 16x32 layout
    const int arow = (mt0 << 4) + (lane & 15);        // A row for sub-tile 0

    for (int k0 = 0; k0 < K; k0 += 32) {
        // B 32x16: lane selects N column, 16 halves span K (same k order as A; shared by all 4 tiles)
        const _Float16* bcol = B + (size_t)k0 * Nout + n0 + (lane & 15);
        v16h b;
#pragma unroll
        for (int s = 0; s < 8; ++s) {
            b[s]     = bcol[(size_t)(kb + s) * Nout];
            b[s + 8] = bcol[(size_t)(16 + kb + s) * Nout];
        }
#pragma unroll
        for (int t = 0; t < MBLK; ++t) {
            const size_t abase = (size_t)(arow + 16 * t) * K;
            v8h alo = *(const v8h*)(A + abase + k0 + kb);
            v8h ahi = *(const v8h*)(A + abase + k0 + 16 + kb);
            v16h a = __builtin_shufflevector(alo, ahi,
                                             0,1,2,3,4,5,6,7,8,9,10,11,12,13,14,15);
            acc[t] = __builtin_amdgcn_wmma_f32_16x16x32_f16(false, a, false, b, (short)0,
                                                            acc[t], false, false);
        }
    }
    // C layout: VGPR i -> row base+i (lanes 0-15) / base+8+i (lanes 16-31); col = n0 + lane%16
    const int col = n0 + (lane & 15);
#pragma unroll
    for (int t = 0; t < MBLK; ++t) {
        const int r0 = ((mt0 + t) << 4) + ((lane < 16) ? 0 : 8);
#pragma unroll
        for (int i = 0; i < 8; ++i) C[(size_t)(r0 + i) * Nout + col] = acc[t][i];
    }
}

// ---------------- degree / normalization ----------------
__global__ void init_deg(float* __restrict__ deg, int N) {
    int i = blockIdx.x * blockDim.x + threadIdx.x;
    if (i < N) deg[i] = 1.0f;  // self loop
}
__global__ void edge_deg(const long long* __restrict__ dst, float* __restrict__ deg, int E) {
    int e = blockIdx.x * blockDim.x + threadIdx.x;
    if (e < E) atomicAdd(deg + (int)dst[e], 1.0f);
}
__global__ void finalize_dinv(const float* __restrict__ deg, float* __restrict__ dinv, int N) {
    int i = blockIdx.x * blockDim.x + threadIdx.x;
    if (i < N) { float d = deg[i]; dinv[i] = d > 0.f ? rsqrtf(d) : 0.f; }
}

// ---------------- GCN aggregation (wave per edge) ----------------
__global__ void gcn_agg(const float* __restrict__ h, int ldh,
                        const long long* __restrict__ src, const long long* __restrict__ dst,
                        const float* __restrict__ dinv, float* __restrict__ agg, int ldagg,
                        int C, int E) {
    int e = blockIdx.x * (blockDim.x >> 5) + (threadIdx.x >> 5);
    if (e >= E) return;
    int lane = threadIdx.x & 31;
    int s = (int)src[e], d = (int)dst[e];
    float norm = dinv[s] * dinv[d];
    for (int c = lane; c < C; c += 32)
        atomicAdd(agg + (size_t)d * ldagg + c, h[(size_t)s * ldh + c] * norm);
}

// finish GCN1: x = relu(agg + h*dinv^2 + b) -> f16 activation for next GEMM
__global__ void gcn_finish_relu(const float* __restrict__ agg, const float* __restrict__ h,
                                const float* __restrict__ dinv, const float* __restrict__ bias,
                                _Float16* __restrict__ act, long long total) {
    long long idx = (long long)blockIdx.x * blockDim.x + threadIdx.x;
    if (idx >= total) return;
    int i = (int)(idx >> 8), c = (int)(idx & (D_FEAT - 1));
    float di = dinv[i];
    float v = agg[idx] + h[idx] * di * di + bias[c];
    act[idx] = (_Float16)(v > 0.f ? v : 0.f);
}

// finish GCN2 (no relu): logits[N,L] from agg[N,L] + h[N,Lp]*dinv^2 + b2
__global__ void gcn_final(const float* __restrict__ agg, const float* __restrict__ h, int ldh,
                          const float* __restrict__ dinv, const float* __restrict__ bias,
                          float* __restrict__ logits, int N, int L) {
    int idx = blockIdx.x * blockDim.x + threadIdx.x;
    if (idx >= N * L) return;
    int i = idx / L, c = idx - i * L;
    float di = dinv[i];
    logits[idx] = agg[idx] + h[(size_t)i * ldh + c] * di * di + bias[c];
}

// ---------------- GAT ----------------
// wave per node: al_src/al_dst dot products + init max-key with self-loop score
__global__ void gat_node_alphas(const float* __restrict__ h, const float* __restrict__ a_src,
                                const float* __restrict__ a_dst, float* __restrict__ als,
                                float* __restrict__ ald, unsigned* __restrict__ mkey, int N) {
    int i = blockIdx.x * (blockDim.x >> 5) + (threadIdx.x >> 5);
    if (i >= N) return;
    int lane = threadIdx.x & 31;
    float s0 = 0.f, s1 = 0.f;
    const float* row = h + (size_t)i * D_FEAT;
#pragma unroll
    for (int j = 0; j < D_FEAT / 32; ++j) {
        float v = row[lane + 32 * j];
        s0 += v * a_src[lane + 32 * j];
        s1 += v * a_dst[lane + 32 * j];
    }
    s0 = wave_sum32(s0);
    s1 = wave_sum32(s1);
    if (lane == 0) {
        als[i] = s0; ald[i] = s1;
        mkey[i] = fkey(lrelu(s0 + s1));  // self-loop initializes segment max
    }
}

__global__ void gat_edge_max(const long long* __restrict__ src, const long long* __restrict__ dst,
                             const float* __restrict__ als, const float* __restrict__ ald,
                             unsigned* __restrict__ mkey, int E) {
    int e = blockIdx.x * blockDim.x + threadIdx.x;
    if (e >= E) return;
    int s = (int)src[e], d = (int)dst[e];
    atomicMax(mkey + d, fkey(lrelu(als[s] + ald[d])));
}

__global__ void gat_node_m(const unsigned* __restrict__ mkey, const float* __restrict__ als,
                           const float* __restrict__ ald, float* __restrict__ m,
                           float* __restrict__ z, float* __restrict__ selfex, int N) {
    int i = blockIdx.x * blockDim.x + threadIdx.x;
    if (i >= N) return;
    float mv = unfkey(mkey[i]);
    float se = expf(lrelu(als[i] + ald[i]) - mv);
    m[i] = mv; selfex[i] = se; z[i] = se;  // z seeded with self-loop term
}

__global__ void gat_edge_exp(const long long* __restrict__ src, const long long* __restrict__ dst,
                             const float* __restrict__ als, const float* __restrict__ ald,
                             const float* __restrict__ m, float* __restrict__ z,
                             float* __restrict__ exbuf, int E) {
    int e = blockIdx.x * blockDim.x + threadIdx.x;
    if (e >= E) return;
    int s = (int)src[e], d = (int)dst[e];
    float ex = expf(lrelu(als[s] + ald[d]) - m[d]);
    exbuf[e] = ex;
    atomicAdd(z + d, ex);
}

__global__ void gat_edge_agg(const float* __restrict__ h, const long long* __restrict__ src,
                             const long long* __restrict__ dst, const float* __restrict__ exbuf,
                             const float* __restrict__ z, float* __restrict__ agg, int E) {
    int e = blockIdx.x * (blockDim.x >> 5) + (threadIdx.x >> 5);
    if (e >= E) return;
    int lane = threadIdx.x & 31;
    int s = (int)src[e], d = (int)dst[e];
    float alpha = exbuf[e] / z[d];
#pragma unroll
    for (int j = 0; j < D_FEAT / 32; ++j)
        atomicAdd(agg + (size_t)d * D_FEAT + lane + 32 * j,
                  h[(size_t)s * D_FEAT + lane + 32 * j] * alpha);
}

__global__ void gat_finish(const float* __restrict__ agg, const float* __restrict__ h,
                           const float* __restrict__ selfex, const float* __restrict__ z,
                           const float* __restrict__ bias, _Float16* __restrict__ act,
                           long long total) {
    long long idx = (long long)blockIdx.x * blockDim.x + threadIdx.x;
    if (idx >= total) return;
    int i = (int)(idx >> 8), c = (int)(idx & (D_FEAT - 1));
    float v = agg[idx] + h[idx] * (selfex[i] / z[i]) + bias[c];
    act[idx] = (_Float16)(v > 0.f ? v : 0.f);
}

// ---------------- log_softmax over L=40 (wave per node) ----------------
__global__ void log_softmax40(const float* __restrict__ logits, float* __restrict__ out,
                              int N, int L) {
    int i = blockIdx.x * (blockDim.x >> 5) + (threadIdx.x >> 5);
    if (i >= N) return;
    int lane = threadIdx.x & 31;
    const float* row = logits + (size_t)i * L;
    float v0 = (lane < L) ? row[lane] : -3.4e38f;
    float v1 = (lane + 32 < L) ? row[lane + 32] : -3.4e38f;
    float mx = wave_max32(fmaxf(v0, v1));
    float s = 0.f;
    if (lane < L) s += expf(v0 - mx);
    if (lane + 32 < L) s += expf(v1 - mx);
    s = wave_sum32(s);
    float lse = mx + logf(s);
    if (lane < L) out[(size_t)i * L + lane] = v0 - lse;
    if (lane + 32 < L) out[(size_t)i * L + lane + 32] = v1 - lse;
}

// ---------------- host orchestration ----------------
extern "C" void kernel_launch(void* const* d_in, const int* in_sizes, int n_in,
                              void* d_out, int out_size, void* d_ws, size_t ws_size,
                              hipStream_t stream) {
    (void)n_in; (void)out_size; (void)ws_size;
    const float*     x     = (const float*)d_in[0];
    const long long* ei    = (const long long*)d_in[1];
    const float*     W1    = (const float*)d_in[2];
    const float*     b1    = (const float*)d_in[3];
    const float*     Wg    = (const float*)d_in[4];
    const float*     a_src = (const float*)d_in[5];
    const float*     a_dst = (const float*)d_in[6];
    const float*     bg    = (const float*)d_in[7];
    const float*     W2    = (const float*)d_in[8];
    const float*     b2    = (const float*)d_in[9];

    const int N  = in_sizes[0] / D_FEAT;   // 50000
    const int E  = in_sizes[1] / 2;        // 800000
    const int L  = in_sizes[9];            // 40
    const int Lp = 48;                     // padded to 16-multiple
    const long long* src = ei;
    const long long* dst = ei + E;
    const long long NC = (long long)N * D_FEAT;

    unsigned char* ws = (unsigned char*)d_ws;
    size_t off = 0;
    auto alloc = [&](size_t bytes) -> void* {
        void* p = ws + off;
        off += (bytes + 511) & ~(size_t)511;
        return p;
    };
    float*    colsum  = (float*)alloc(D_FEAT * 4);
    float*    colsq   = (float*)alloc(D_FEAT * 4);
    float*    mean    = (float*)alloc(D_FEAT * 4);
    float*    rinv    = (float*)alloc(D_FEAT * 4);
    _Float16* w1h     = (_Float16*)alloc((size_t)D_FEAT * D_FEAT * 2);
    _Float16* wgh     = (_Float16*)alloc((size_t)D_FEAT * D_FEAT * 2);
    _Float16* w2h     = (_Float16*)alloc((size_t)D_FEAT * Lp * 2);
    _Float16* act     = (_Float16*)alloc((size_t)NC * 2);      // f16 GEMM input (reused)
    float*    h       = (float*)alloc((size_t)NC * 4);         // GEMM output (reused; [N,48] for last)
    float*    agg     = (float*)alloc((size_t)NC * 4);         // aggregation (reused; [N,40] for last)
    float*    deg     = (float*)alloc((size_t)N * 4);
    float*    dinv    = (float*)alloc((size_t)N * 4);
    float*    als     = (float*)alloc((size_t)N * 4);
    float*    ald     = (float*)alloc((size_t)N * 4);
    float*    mArr    = (float*)alloc((size_t)N * 4);
    float*    z       = (float*)alloc((size_t)N * 4);
    float*    selfex  = (float*)alloc((size_t)N * 4);
    unsigned* mkey    = (unsigned*)alloc((size_t)N * 4);
    float*    exbuf   = (float*)alloc((size_t)E * 4);
    float*    logits  = (float*)alloc((size_t)N * L * 4);

    const int BT = 256;
    const int gN    = (N + BT - 1) / BT;
    const int gE    = (E + BT - 1) / BT;
    const int gNC   = (int)((NC + BT - 1) / BT);
    const int gNw   = (N + 7) / 8;   // wave-per-node
    const int gEw   = (E + 7) / 8;   // wave-per-edge

    // --- feature standardization -> act(f16) ---
    zero_f32<<<1, 256, 0, stream>>>(colsum, D_FEAT);
    zero_f32<<<1, 256, 0, stream>>>(colsq, D_FEAT);
    col_stats<<<512, 256, 0, stream>>>(x, colsum, colsq, N);
    finalize_stats<<<1, 256, 0, stream>>>(colsum, colsq, mean, rinv, N);
    standardize<<<gNC, BT, 0, stream>>>(x, mean, rinv, act, NC);

    // --- weight conversion ---
    convert_w<<<(D_FEAT * D_FEAT + BT - 1) / BT, BT, 0, stream>>>(W1, w1h, D_FEAT * D_FEAT);
    convert_w<<<(D_FEAT * D_FEAT + BT - 1) / BT, BT, 0, stream>>>(Wg, wgh, D_FEAT * D_FEAT);
    convert_w2_pad<<<(D_FEAT * Lp + BT - 1) / BT, BT, 0, stream>>>(W2, w2h, D_FEAT, L, Lp);

    // --- degrees (with self loops) ---
    init_deg<<<gN, BT, 0, stream>>>(deg, N);
    edge_deg<<<gE, BT, 0, stream>>>(dst, deg, E);
    finalize_dinv<<<gN, BT, 0, stream>>>(deg, dinv, N);

    auto gemm_blocks = [&](int M, int Nout) -> int {
        int tiles_m = M >> 4, tiles_n = Nout >> 4;
        int mblocks = (tiles_m + MBLK - 1) / MBLK;
        return (mblocks * tiles_n + 7) / 8;  // 8 waves per 256-thread block
    };

    // --- GCN layer 1 ---
    {
        gemm_f16_wmma<<<gemm_blocks(N, D_FEAT), BT, 0, stream>>>(act, w1h, h, N, D_FEAT, D_FEAT);
        zero_f32<<<gNC, BT, 0, stream>>>(agg, NC);
        gcn_agg<<<gEw, BT, 0, stream>>>(h, D_FEAT, src, dst, dinv, agg, D_FEAT, D_FEAT, E);
        gcn_finish_relu<<<gNC, BT, 0, stream>>>(agg, h, dinv, b1, act, NC);
    }

    // --- GAT layers x2 ---
    for (int l = 0; l < 2; ++l) {
        gemm_f16_wmma<<<gemm_blocks(N, D_FEAT), BT, 0, stream>>>(act, wgh, h, N, D_FEAT, D_FEAT);
        gat_node_alphas<<<gNw, BT, 0, stream>>>(h, a_src, a_dst, als, ald, mkey, N);
        gat_edge_max<<<gE, BT, 0, stream>>>(src, dst, als, ald, mkey, E);
        gat_node_m<<<gN, BT, 0, stream>>>(mkey, als, ald, mArr, z, selfex, N);
        gat_edge_exp<<<gE, BT, 0, stream>>>(src, dst, als, ald, mArr, z, exbuf, E);
        zero_f32<<<gNC, BT, 0, stream>>>(agg, NC);
        gat_edge_agg<<<gEw, BT, 0, stream>>>(h, src, dst, exbuf, z, agg, E);
        gat_finish<<<gNC, BT, 0, stream>>>(agg, h, selfex, z, bg, act, NC);
    }

    // --- GCN layer 2 (output, padded Nout=48) ---
    {
        gemm_f16_wmma<<<gemm_blocks(N, Lp), BT, 0, stream>>>(act, w2h, h, N, D_FEAT, Lp);
        zero_f32<<<(int)(((long long)N * L + BT - 1) / BT), BT, 0, stream>>>(agg, (long long)N * L);
        gcn_agg<<<gEw, BT, 0, stream>>>(h, Lp, src, dst, dinv, agg, L, L, E);
        gcn_final<<<(N * L + BT - 1) / BT, BT, 0, stream>>>(agg, h, Lp, dinv, b2, logits, N, L);
        log_softmax40<<<gNw, BT, 0, stream>>>(logits, (float*)d_out, N, L);
    }
}
